// SolarRingModel_74096775790763
// MI455X (gfx1250) — matrix operand order repaired and checked by reading, not compiled, not run on hardware
//
#include <hip/hip_runtime.h>
#include <hip/hip_bf16.h>

// ---------------- model constants ----------------
#define DM   512          // D_MODEL
#define VV   32000        // vocab
#define LL   8            // layers
#define KS   16           // ring slots
#define BB   4
#define TT   1024
#define NR   4
#define NS   2

typedef __attribute__((ext_vector_type(16))) _Float16 v16h;
typedef __attribute__((ext_vector_type(8)))  _Float16 v8h;
typedef __attribute__((ext_vector_type(8)))  float    v8f;
typedef unsigned int u32x4 __attribute__((ext_vector_type(4)));
typedef int          i32x8 __attribute__((ext_vector_type(8)));
typedef int          i32x4 __attribute__((ext_vector_type(4)));

// ---------------- kernel 1: emb fp32 -> fp16 ----------------
__global__ __launch_bounds__(256) void emb_to_f16_kernel(const float* __restrict__ emb,
                                                         _Float16* __restrict__ emb16,
                                                         long n) {
    long i = (long)blockIdx.x * blockDim.x + threadIdx.x;
    if (i < n) emb16[i] = (_Float16)emb[i];
}

// ---------------- kernel 2: recurrent ring scan ----------------
// grid = B, block = 512 (16 waves).  mem lives in LDS for the whole scan.
__global__ __launch_bounds__(512) void recurrent_kernel(
    const int*   __restrict__ token_ids,
    const float* __restrict__ emb,
    const float* __restrict__ Wq,
    const float* __restrict__ Wv,
    const float* __restrict__ ln_scale,
    const float* __restrict__ ln_bias,
    const float* __restrict__ W_role,
    const float* __restrict__ W_spawn,
    const float* __restrict__ W_write,
    const float* __restrict__ W_skip,
    float* __restrict__ xs_out,      // [B,T,D]
    float* __restrict__ mem_out,     // [B,KS,D]
    float* __restrict__ roles_out,   // [B,T,NR]
    float* __restrict__ spawns_out)  // [B,T,NS]
{
    __shared__ float mem[KS * DM];   // 32 KB ring memory
    __shared__ float x[DM];
    __shared__ float buf[DM];        // q, then read-vector
    __shared__ float skip[DM];
    __shared__ float red1[16];
    __shared__ float red2[16];
    __shared__ float attn[KS];
    __shared__ float stats[2];
    __shared__ float role_acc[NR];
    __shared__ float spawn_acc[NS];

    const int b    = blockIdx.x;
    const int tid  = threadIdx.x;        // 0..511
    const int lane = tid & 31;
    const int wave = tid >> 5;           // 0..15

    for (int i = tid; i < KS * DM; i += 512) mem[i] = 0.f;
    __syncthreads();

    for (int t = 0; t < TT; ++t) {
        const int tok = token_ids[b * TT + t];
        x[tid] = emb[(long)tok * DM + tid];
        if (tid < NR) role_acc[tid] = 0.f;
        if (tid < NS) spawn_acc[tid] = 0.f;
        __syncthreads();

        for (int i = 0; i < LL; ++i) {
            const float* wq = Wq + (long)i * DM * DM;
            // q = x @ Wq[i]
            {
                float acc = 0.f;
                #pragma unroll 4
                for (int k = 0; k < DM; ++k) acc = fmaf(x[k], wq[(long)k * DM + tid], acc);
                buf[tid] = acc;
            }
            __syncthreads();
            // scores: wave j computes mem[j] . q
            {
                const float* mrow = mem + wave * DM;
                float p = 0.f;
                for (int k = lane; k < DM; k += 32) p = fmaf(mrow[k], buf[k], p);
                for (int o = 16; o; o >>= 1) p += __shfl_down(p, o);
                if (lane == 0) red1[wave] = p;
            }
            __syncthreads();
            if (tid == 0) {   // softmax over 16 scores
                float mx = red1[0];
                for (int j = 1; j < KS; ++j) mx = fmaxf(mx, red1[j]);
                float s = 0.f;
                for (int j = 0; j < KS; ++j) { float e = __expf(red1[j] - mx); attn[j] = e; s += e; }
                float inv = 1.f / s;
                for (int j = 0; j < KS; ++j) attn[j] *= inv;
            }
            __syncthreads();
            // read = attn @ mem
            {
                float rd = 0.f;
                #pragma unroll
                for (int j = 0; j < KS; ++j) rd = fmaf(attn[j], mem[j * DM + tid], rd);
                __syncthreads();
                buf[tid] = rd;
            }
            __syncthreads();
            // v = read @ Wv[i];  xr = x + v;  LN
            float xr;
            {
                const float* wv = Wv + (long)i * DM * DM;
                float acc = 0.f;
                #pragma unroll 4
                for (int k = 0; k < DM; ++k) acc = fmaf(buf[k], wv[(long)k * DM + tid], acc);
                xr = x[tid] + acc;
            }
            {
                float s1 = xr, s2 = xr * xr;
                for (int o = 16; o; o >>= 1) { s1 += __shfl_down(s1, o); s2 += __shfl_down(s2, o); }
                if (lane == 0) { red1[wave] = s1; red2[wave] = s2; }
                __syncthreads();
                if (tid == 0) {
                    float a = 0.f, q2 = 0.f;
                    for (int j = 0; j < 16; ++j) { a += red1[j]; q2 += red2[j]; }
                    float mu  = a * (1.f / DM);
                    float var = q2 * (1.f / DM) - mu * mu;
                    stats[0] = mu;
                    stats[1] = rsqrtf(var + 1e-5f);
                }
                __syncthreads();
            }
            float xn = (xr - stats[0]) * stats[1] * ln_scale[i * DM + tid] + ln_bias[i * DM + tid];
            x[tid] = xn;
            __syncthreads();

            // roles (waves 0..3) and spawns (waves 4..5): dot(x, column)
            if (wave < NR) {
                const float* wr = W_role + (long)i * DM * NR;
                float p = 0.f;
                for (int k = lane; k < DM; k += 32) p = fmaf(x[k], wr[k * NR + wave], p);
                for (int o = 16; o; o >>= 1) p += __shfl_down(p, o);
                if (lane == 0) role_acc[wave] += p;
            } else if (wave < NR + NS) {
                const int c = wave - NR;
                const float* wsp = W_spawn + (long)i * DM * NS;
                float p = 0.f;
                for (int k = lane; k < DM; k += 32) p = fmaf(x[k], wsp[k * NS + c], p);
                for (int o = 16; o; o >>= 1) p += __shfl_down(p, o);
                if (lane == 0) spawn_acc[c] += p;
            }

            if (i == 0) {   // ring write + skip capture (post-LN x)
                float acc = 0.f;
                #pragma unroll 4
                for (int k = 0; k < DM; ++k) acc = fmaf(x[k], W_write[(long)k * DM + tid], acc);
                mem[(t & (KS - 1)) * DM + tid] = acc;
                skip[tid] = x[tid];
            }
            __syncthreads();
        }

        // final skip connection and outputs
        {
            float acc = 0.f;
            #pragma unroll 4
            for (int k = 0; k < DM; ++k) acc = fmaf(skip[k], W_skip[(long)k * DM + tid], acc);
            const float xf = x[tid] + acc;
            xs_out[((long)b * TT + t) * DM + tid] = xf;
        }
        if (tid < NR) roles_out[((long)b * TT + t) * NR + tid] = role_acc[tid] * (1.f / LL);
        if (tid < NS) spawns_out[((long)b * TT + t) * NS + tid] = spawn_acc[tid] * (1.f / LL);
        __syncthreads();
    }

    for (int i = tid; i < KS * DM; i += 512) mem_out[(long)b * KS * DM + i] = mem[i];
}

// ---------------- kernel 3: mem_vec = LN(mem.flat @ W_out + b_out) ----------------
__global__ __launch_bounds__(512) void memvec_kernel(
    const float* __restrict__ mem_ws,    // [B,KS*D]
    const float* __restrict__ W_out,     // [KS*D, D]
    const float* __restrict__ b_out,
    const float* __restrict__ on_scale,
    const float* __restrict__ on_bias,
    float* __restrict__ memvec_ws,       // [B,D]
    float* __restrict__ ctx_out)         // [B,D]
{
    __shared__ float red1[16];
    __shared__ float red2[16];
    __shared__ float stats[2];
    const int b = blockIdx.x, tid = threadIdx.x;
    const int lane = tid & 31, wave = tid >> 5;
    const float* m = mem_ws + (long)b * KS * DM;

    float acc = b_out[tid];
    #pragma unroll 4
    for (int k = 0; k < KS * DM; ++k) acc = fmaf(m[k], W_out[(long)k * DM + tid], acc);

    float s1 = acc, s2 = acc * acc;
    for (int o = 16; o; o >>= 1) { s1 += __shfl_down(s1, o); s2 += __shfl_down(s2, o); }
    if (lane == 0) { red1[wave] = s1; red2[wave] = s2; }
    __syncthreads();
    if (tid == 0) {
        float a = 0.f, q2 = 0.f;
        for (int j = 0; j < 16; ++j) { a += red1[j]; q2 += red2[j]; }
        float mu  = a * (1.f / DM);
        float var = q2 * (1.f / DM) - mu * mu;
        stats[0] = mu; stats[1] = rsqrtf(var + 1e-5f);
    }
    __syncthreads();
    float v = (acc - stats[0]) * stats[1] * on_scale[tid] + on_bias[tid];
    memvec_ws[(long)b * DM + tid] = v;
    ctx_out[(long)b * DM + tid]   = v;
}

// ---------------- kernel 4: y16 = f16(xs + mem_vec[b]) ----------------
__global__ __launch_bounds__(256) void y16_kernel(const float* __restrict__ xs,
                                                  const float* __restrict__ memvec,
                                                  _Float16* __restrict__ y16) {
    long i = (long)blockIdx.x * blockDim.x + threadIdx.x;   // over B*T*D
    const int b = (int)(i >> 19);              // T*D = 524288 = 2^19
    const int d = (int)(i & (DM - 1));
    y16[i] = (_Float16)(xs[i] + memvec[b * DM + d]);
}

// ---------------- kernel 5: logits = y @ emb16^T via WMMA + TDM ----------------
// grid.x = N/256 = 125, grid.y = M/16 = 256.  block = 128 (4 waves).
// A-tile (16 x 512 f16) is DMA'd once per block into LDS by the Tensor Data
// Mover with 16B padding every 1024B (row stride 1040B -> no LDS bank
// conflicts across the 16 lanes of an A-fragment read).  Waves then read A
// from LDS (ds_load_b128) and stream B (emb16, L2-resident) from global.
#define APAD 8                           // 8 halves = 16 B pad per 512-half row
__global__ __launch_bounds__(128) void logits_wmma_kernel(
    const _Float16* __restrict__ y16,     // [M=B*T, D]
    const _Float16* __restrict__ emb16,   // [V, D]  (B operand = emb.T)
    float* __restrict__ logits)           // [M, V]
{
    __shared__ _Float16 Atile[16][DM + APAD];   // 16.6 KB

    const int tid  = threadIdx.x;
    const int lane = tid & 31;
    const int wave = tid >> 5;
    const int m0   = blockIdx.y * 16;
    const int n0   = blockIdx.x * 256 + wave * 64;
    const int hsel = lane >> 4;           // 0 or 1
    const int lmod = lane & 15;

    // ---- TDM: DMA the 16x512 f16 A tile into LDS (one wave issues) ----
    if (tid < 32) {
        const unsigned lds_base = (unsigned)(unsigned long long)&Atile[0][0];
        const unsigned long long ga = (unsigned long long)(const void*)(y16 + (long)m0 * DM);
        u32x4 g0;
        g0[0] = 1u;                                   // count=1, user desc
        g0[1] = lds_base;                             // lds_addr
        g0[2] = (unsigned)(ga & 0xffffffffu);         // global_addr[31:0]
        g0[3] = (unsigned)((ga >> 32) & 0x01ffffffu)  // global_addr[56:32]
              | (2u << 30);                           // type=2 (image)
        i32x8 g1;
        g1[0] = 0x07D10000;     // data_size=2B, pad_enable, interval=256DW, amount=4DW
        g1[1] = (int)(512u << 16);                    // tensor_dim0 = 512 (lo16)
        g1[2] = (int)(4096u << 16);                   // dim0 hi=0 | tensor_dim1=4096 lo16
        g1[3] = (int)(512u << 16);                    // dim1 hi=0 | tile_dim0=512
        g1[4] = 16;                                   // tile_dim1=16, tile_dim2=0
        g1[5] = 512;                                  // tensor_dim0_stride = 512
        g1[6] = 0;
        g1[7] = 0;
        i32x4 z4 = {};
        i32x8 z8 = {};
        __builtin_amdgcn_tensor_load_to_lds(g0, g1, z4, z4, z8, 0);
        __builtin_amdgcn_s_wait_tensorcnt(0);
    }
    __syncthreads();

    v8f c0 = {}, c1 = {}, c2 = {}, c3 = {};
    const _Float16* arow = &Atile[lmod][0];

    for (int k0 = 0; k0 < DM; k0 += 32) {
        // A fragment from LDS: lanes 0-15 K 0-15; lanes 16-31 K 16-31
        v8h a_lo = *(const v8h*)(arow + k0 + hsel * 16);
        v8h a_hi = *(const v8h*)(arow + k0 + hsel * 16 + 8);
        v16h a;
        #pragma unroll
        for (int z = 0; z < 8; ++z) { a[z] = a_lo[z]; a[8 + z] = a_hi[z]; }

        // B fragments: lane = column n; lanes 0-15 K 0-15, lanes 16-31 K 16-31
        const long koff = k0 + hsel * 16;
        const _Float16* bbase = emb16 + (long)(n0 + lmod) * DM + koff;
        v16h b0 = *(const v16h*)(bbase);
        v16h b1 = *(const v16h*)(bbase + 16l * DM);
        v16h b2 = *(const v16h*)(bbase + 32l * DM);
        v16h b3 = *(const v16h*)(bbase + 48l * DM);
        __builtin_prefetch(bbase + 32, 0, 3);

        c0 = __builtin_amdgcn_wmma_f32_16x16x32_f16(false, a, false, b0, (short)0, c0, false, false);
        c1 = __builtin_amdgcn_wmma_f32_16x16x32_f16(false, a, false, b1, (short)0, c1, false, false);
        c2 = __builtin_amdgcn_wmma_f32_16x16x32_f16(false, a, false, b2, (short)0, c2, false, false);
        c3 = __builtin_amdgcn_wmma_f32_16x16x32_f16(false, a, false, b3, (short)0, c3, false, false);
    }

    // C/D layout: VGPR r -> row m0 + r + 8*(lane>>4), col = n-base + (lane&15)
    const int rowb = m0 + hsel * 8;
    #pragma unroll
    for (int r = 0; r < 8; ++r) {
        float* dst = logits + (long)(rowb + r) * VV + n0 + lmod;
        dst[0]  = c0[r];
        dst[16] = c1[r];
        dst[32] = c2[r];
        dst[48] = c3[r];
    }
}

// ---------------- launcher ----------------
extern "C" void kernel_launch(void* const* d_in, const int* in_sizes, int n_in,
                              void* d_out, int out_size, void* d_ws, size_t ws_size,
                              hipStream_t stream) {
    const int*   token_ids = (const int*)  d_in[0];
    const float* emb       = (const float*)d_in[1];
    const float* Wq        = (const float*)d_in[2];
    const float* Wv        = (const float*)d_in[3];
    const float* ln_scale  = (const float*)d_in[4];
    const float* ln_bias   = (const float*)d_in[5];
    const float* W_role    = (const float*)d_in[6];
    const float* W_spawn   = (const float*)d_in[7];
    const float* W_write   = (const float*)d_in[8];
    const float* W_skip    = (const float*)d_in[9];
    const float* W_out     = (const float*)d_in[10];
    const float* b_out     = (const float*)d_in[11];
    const float* on_scale  = (const float*)d_in[12];
    const float* on_bias   = (const float*)d_in[13];

    // output regions: logits, roles, spawns, ctx (flat concat, fp32)
    float* logits = (float*)d_out;
    float* roles  = logits + (size_t)BB * TT * VV;
    float* spawns = roles  + (size_t)BB * TT * NR;
    float* ctx    = spawns + (size_t)BB * TT * NS;

    // workspace layout
    char* ws = (char*)d_ws;
    const size_t emb16_bytes  = (size_t)VV * DM * sizeof(_Float16);   // 32.77 MB
    const size_t xs_bytes     = (size_t)BB * TT * DM * sizeof(float); //  8.39 MB
    const size_t mem_bytes    = (size_t)BB * KS * DM * sizeof(float); //  128 KB
    const size_t memvec_bytes = (size_t)BB * DM * sizeof(float);      //    8 KB

    _Float16* emb16  = (_Float16*)ws;
    float*    xs     = (float*)(ws + emb16_bytes);
    float*    mem_ws = (float*)(ws + emb16_bytes + xs_bytes);
    float*    memvec = (float*)(ws + emb16_bytes + xs_bytes + mem_bytes);
    _Float16* y16    = (_Float16*)(ws + emb16_bytes + xs_bytes + mem_bytes + memvec_bytes);

    // 1. emb -> f16 (B operand of tied lm_head, stays L2-resident: 33 MB)
    {
        long n = (long)VV * DM;
        emb_to_f16_kernel<<<dim3((unsigned)((n + 255) / 256)), dim3(256), 0, stream>>>(emb, emb16, n);
    }
    // 2. sequential ring scan: one WGP-resident block per sequence
    recurrent_kernel<<<dim3(BB), dim3(512), 0, stream>>>(
        token_ids, emb, Wq, Wv, ln_scale, ln_bias, W_role, W_spawn,
        W_write, W_skip, xs, mem_ws, roles, spawns);
    // 3. mem_vec + ctx
    memvec_kernel<<<dim3(BB), dim3(512), 0, stream>>>(
        mem_ws, W_out, b_out, on_scale, on_bias, memvec, ctx);
    // 4. y16 = f16(xs + mem_vec)
    {
        long n = (long)BB * TT * DM;
        y16_kernel<<<dim3((unsigned)(n / 256)), dim3(256), 0, stream>>>(xs, memvec, y16);
    }
    // 5. logits GEMM (134 GFLOP, 0.52 GB store — the roofline term) via WMMA f16
    logits_wmma_kernel<<<dim3(VV / 256, (BB * TT) / 16), dim3(128), 0, stream>>>(y16, emb16, logits);
}
